// Transformer_Block_62242666053728
// MI455X (gfx1250) — matrix operand order repaired and checked by reading, not compiled, not run on hardware
//
#include <hip/hip_runtime.h>
#include <hip/hip_bf16.h>

typedef __attribute__((ext_vector_type(8)))  float  v8f;
typedef __attribute__((ext_vector_type(16))) __bf16 v16bf;

#define N_NODES 10000
#define N_EDGES 640000
#define CH 128

union ABU {
  unsigned int u[8];
  v16bf h;
};

// ---------------------------------------------------------------------------
// Kernel 1: build bf16 mixed-precision copies (x3h = bf16(x*W3), xh = bf16(x))
// and zero the output accumulator.
// ---------------------------------------------------------------------------
__global__ void __launch_bounds__(256)
prep_kernel(const float* __restrict__ x,
            const float* __restrict__ W3,
            __bf16* __restrict__ x3h,
            __bf16* __restrict__ xh,
            float* __restrict__ out) {
  int i = blockIdx.x * blockDim.x + threadIdx.x;
  if (i < N_NODES * CH) {
    float xv = x[i];
    x3h[i] = (__bf16)(xv * W3[i & (CH - 1)]);
    xh[i]  = (__bf16)xv;
    out[i] = 0.0f;
  }
}

// ---------------------------------------------------------------------------
// Kernel 2: per-edge attention dots via v_wmma_f32_16x16x32_bf16.
// One wave (32 lanes) handles 16 edges: D[i][j] = <x3h[src_i], xh[dst_j]>,
// accumulated over K=128 in four K=32 chunks; we keep the diagonal.
// A-matrix 16x32 bf16 layout: lane l -> row M=l%16;
//   lanes<16:  VGPR0-3 = K 0..7,  VGPR4-7 = K 16..23
//   lanes>=16: VGPR0-3 = K 8..15, VGPR4-7 = K 24..31
// B-matrix 32x16 bf16 layout: lane l -> col N=l%16;
//   lanes<16:  VGPR0-3 = K 0..7, VGPR4-7 = K 16..23 ; lanes>=16: K 8..15, K 24..31
// Both are 16B-contiguous per lane -> global_load_b128 pairs.
// ---------------------------------------------------------------------------
__global__ void __launch_bounds__(128)
attn_kernel(const __bf16* __restrict__ x3h,
            const __bf16* __restrict__ xh,
            const int* __restrict__ srcI,
            const int* __restrict__ dstI,
            float* __restrict__ attn) {
  const int lane  = threadIdx.x & 31;
  const int wave  = threadIdx.x >> 5;
  const int tile  = blockIdx.x * 4 + wave;   // 16 edges per tile
  const int eBase = tile * 16;

  const int m = lane & 15;
  const int e = eBase + m;
  const int s = srcI[e];
  const int d = dstI[e];
  const __bf16* __restrict__ rowA = x3h + (size_t)s * CH;
  const __bf16* __restrict__ rowB = xh  + (size_t)d * CH;

  const int hi = lane >> 4;   // 0 or 1
  const int ko = hi << 3;     // A half-offset: 0 / 8
  const int kb = hi << 3;     // B half-offset: 0 / 8

  v8f acc = {};
#pragma unroll
  for (int k0 = 0; k0 < CH; k0 += 32) {
    ABU a, b;
    uint4 a0 = *(const uint4*)(rowA + k0 + ko);
    uint4 a1 = *(const uint4*)(rowA + k0 + 16 + ko);
    uint4 b0 = *(const uint4*)(rowB + k0 + kb);
    uint4 b1 = *(const uint4*)(rowB + k0 + 16 + kb);
    a.u[0] = a0.x; a.u[1] = a0.y; a.u[2] = a0.z; a.u[3] = a0.w;
    a.u[4] = a1.x; a.u[5] = a1.y; a.u[6] = a1.z; a.u[7] = a1.w;
    b.u[0] = b0.x; b.u[1] = b0.y; b.u[2] = b0.z; b.u[3] = b0.w;
    b.u[4] = b1.x; b.u[5] = b1.y; b.u[6] = b1.z; b.u[7] = b1.w;
    acc = __builtin_amdgcn_wmma_f32_16x16x32_bf16(
        /*neg_a=*/false, a.h, /*neg_b=*/false, b.h,
        /*c_mod=*/(short)0, acc, /*reuse_a=*/false, /*reuse_b=*/false);
  }

  // Diagonal D[i][i]: lanes 0-7 hold M=N=lane in VGPR 'lane';
  // lanes 24-31 hold M=N=lane-16 in VGPR 'lane-24'.
  const int sel = (lane < 8) ? lane : (lane - 24);
  float v = acc[0];
#pragma unroll
  for (int r = 1; r < 8; ++r) v = (sel == r) ? acc[r] : v;

  if (lane < 8)        attn[eBase + lane]      = v;
  else if (lane >= 24) attn[eBase + lane - 16] = v;
}

// ---------------------------------------------------------------------------
// Kernel 3: scatter  out[dst] += attn * (x[src] * W2)
// One wave per edge, 4 channels per lane, hardware f32 atomics into L2.
// ---------------------------------------------------------------------------
__global__ void __launch_bounds__(256)
scatter_kernel(const float* __restrict__ x,
               const float* __restrict__ W2,
               const int* __restrict__ srcI,
               const int* __restrict__ dstI,
               const float* __restrict__ attn,
               float* __restrict__ out) {
  const int lane = threadIdx.x & 31;
  const int wave = threadIdx.x >> 5;
  const int e    = blockIdx.x * 8 + wave;

  const float a = attn[e];
  const int   s = srcI[e];
  const int   d = dstI[e];
  const int   c = lane * 4;

  float4 xs = *(const float4*)(x  + (size_t)s * CH + c);
  float4 w  = *(const float4*)(W2 + c);
  float* o  = out + (size_t)d * CH + c;

  __hip_atomic_fetch_add(o + 0, a * xs.x * w.x, __ATOMIC_RELAXED, __HIP_MEMORY_SCOPE_AGENT);
  __hip_atomic_fetch_add(o + 1, a * xs.y * w.y, __ATOMIC_RELAXED, __HIP_MEMORY_SCOPE_AGENT);
  __hip_atomic_fetch_add(o + 2, a * xs.z * w.z, __ATOMIC_RELAXED, __HIP_MEMORY_SCOPE_AGENT);
  __hip_atomic_fetch_add(o + 3, a * xs.w * w.w, __ATOMIC_RELAXED, __HIP_MEMORY_SCOPE_AGENT);
}

// ---------------------------------------------------------------------------
extern "C" void kernel_launch(void* const* d_in, const int* in_sizes, int n_in,
                              void* d_out, int out_size, void* d_ws, size_t ws_size,
                              hipStream_t stream) {
  const float* x    = (const float*)d_in[0];
  const int*   eidx = (const int*)d_in[1];     // [2, E] edge_index
  const float* W2   = (const float*)d_in[2];
  const float* W3   = (const float*)d_in[3];
  float*       out  = (float*)d_out;

  // workspace layout: attn[E] f32 | x3h[N*C] bf16 | xh[N*C] bf16  (~7.7 MB)
  float*  attn = (float*)d_ws;
  __bf16* x3h  = (__bf16*)((char*)d_ws + (size_t)N_EDGES * 4);
  __bf16* xh   = (__bf16*)((char*)d_ws + (size_t)N_EDGES * 4 + (size_t)N_NODES * CH * 2);

  const int* srcI = eidx;             // edge_index[0]
  const int* dstI = eidx + N_EDGES;   // edge_index[1]

  prep_kernel<<<(N_NODES * CH + 255) / 256, 256, 0, stream>>>(x, W3, x3h, xh, out);
  attn_kernel<<<N_EDGES / 64, 128, 0, stream>>>(x3h, xh, srcI, dstI, attn);
  scatter_kernel<<<N_EDGES / 8, 256, 0, stream>>>(x, W2, srcI, dstI, attn, out);
}
